// GATNet_824633721351
// MI455X (gfx1250) — matrix operand order repaired and checked by reading, not compiled, not run on hardware
//
#include <hip/hip_runtime.h>
#include <hip/hip_bf16.h>

// ---------------- WMMA types ----------------
typedef __attribute__((ext_vector_type(16))) __bf16 bf16x16;
typedef __attribute__((ext_vector_type(8)))  float  f32x8;

struct U4x2 { uint4 a, b; };   // 32 bytes == 16 bf16 == one WMMA A/B fragment

// ---------------- constants ----------------
#define NN      20000      // nodes (= 16*1250 = 32*625)
#define GG      128        // graphs
#define F_IN    78
#define K1P     96         // F_IN padded to multiple of 32
#define H1      10         // heads layer 1
#define CC      128        // out channels per head
#define F1      (H1*CC)    // 1280
#define NEG_SLOPE 0.2f
#define ENC_NEG_INF 0x007FFFFFu   // order-preserving encoding of -inf

__device__ __forceinline__ unsigned short f32_to_bf16(float f) {
  unsigned u = __builtin_bit_cast(unsigned, f);
  unsigned r = 0x7FFFu + ((u >> 16) & 1u);
  return (unsigned short)((u + r) >> 16);
}
__device__ __forceinline__ unsigned enc_f32(float f) {
  unsigned u = __builtin_bit_cast(unsigned, f);
  return (u & 0x80000000u) ? ~u : (u | 0x80000000u);
}
__device__ __forceinline__ float dec_f32(unsigned e) {
  unsigned u = (e & 0x80000000u) ? (e & 0x7FFFFFFFu) : ~e;
  return __builtin_bit_cast(float, u);
}
__device__ __forceinline__ void get_edge(const int* ei, int E, int e, int& s, int& d) {
  if (e < E) { s = ei[e]; d = ei[E + e]; } else { s = e - E; d = e - E; }  // self loops appended
}

// ---------------- fill ----------------
__global__ void fill_u32_k(unsigned* p, unsigned v, long n) {
  long i = (long)blockIdx.x * blockDim.x + threadIdx.x;
  if (i < n) p[i] = v;
}

// ---------------- convert x -> bf16, pad K to 96 ----------------
__global__ void cvt_x_k(const float* x, unsigned short* xp) {
  long i = (long)blockIdx.x * blockDim.x + threadIdx.x;
  if (i >= (long)NN * K1P) return;
  int n = (int)(i / K1P), k = (int)(i % K1P);
  xp[i] = (k < F_IN) ? f32_to_bf16(x[(long)n * F_IN + k]) : (unsigned short)0;
}

// ------ pack W[K,N] (f32) into per-lane WMMA B-fragment order (bf16) ------
// packed index = ((kt*nNt + nt)*32 + lane)*16 + i ; K = kt*32 + (lane>>4)*16 + i ; Ncol = nt*16 + (lane&15)
__global__ void pack_w_k(const float* W, unsigned short* Wp, int Korig, int Kpad, int Ncols) {
  long i = (long)blockIdx.x * blockDim.x + threadIdx.x;
  long total = (long)Kpad * Ncols;
  if (i >= total) return;
  int elem = (int)(i & 15);
  long t = i >> 4;
  int lane = (int)(t & 31); t >>= 5;
  int nNt = Ncols / 16;
  int nt = (int)(t % nNt);
  int kt = (int)(t / nNt);
  int K = kt * 32 + (lane >> 4) * 16 + elem;
  int Ncol = nt * 16 + (lane & 15);
  Wp[i] = (K < Korig) ? f32_to_bf16(W[(long)K * Ncols + Ncol]) : (unsigned short)0;
}

// ---------------- BF16 WMMA GEMM: C[M,N] = A[M,K] * B[K,N] ----------------
// block = 256 (8 waves); wave -> TWO 16x16 C tiles along M sharing one B fragment.
// grid = (M/32, Ntiles/8). 2 WMMAs per 6 b128 loads per K-step.
__global__ void gemm_bf16_wmma_k(const unsigned short* __restrict__ A,
                                 const unsigned short* __restrict__ Bp,
                                 float* __restrict__ C,
                                 int lda, int ldc, int n_tiles, int k_tiles) {
  int wave = threadIdx.x >> 5;
  int lane = threadIdx.x & 31;
  int mt0 = blockIdx.x * 2;                      // two consecutive M tiles
  int nt = blockIdx.y * 8 + wave;
  if (nt >= n_tiles) return;                     // wave-uniform
  int grp = lane >> 4;
  int mlane = lane & 15;
  const unsigned short* arow0 = A + (size_t)(mt0 * 16 + mlane) * lda + grp * 16;
  const unsigned short* arow1 = arow0 + (size_t)16 * lda;
  const unsigned short* bptr  = Bp + ((size_t)nt * 32 + lane) * 16;
  size_t bstep = (size_t)n_tiles * 512;          // next K-tile block
  f32x8 acc0 = {}, acc1 = {};
  for (int kt = 0; kt < k_tiles; ++kt) {
    U4x2 bv, a0v, a1v;
    bv.a  = *reinterpret_cast<const uint4*>(bptr);
    bv.b  = *reinterpret_cast<const uint4*>(bptr + 8);
    a0v.a = *reinterpret_cast<const uint4*>(arow0 + kt * 32);
    a0v.b = *reinterpret_cast<const uint4*>(arow0 + kt * 32 + 8);
    a1v.a = *reinterpret_cast<const uint4*>(arow1 + kt * 32);
    a1v.b = *reinterpret_cast<const uint4*>(arow1 + kt * 32 + 8);
    bf16x16 bf = __builtin_bit_cast(bf16x16, bv);
    bf16x16 a0 = __builtin_bit_cast(bf16x16, a0v);
    bf16x16 a1 = __builtin_bit_cast(bf16x16, a1v);
    acc0 = __builtin_amdgcn_wmma_f32_16x16x32_bf16(false, a0, false, bf,
                                                   (short)0, acc0, false, false);
    acc1 = __builtin_amdgcn_wmma_f32_16x16x32_bf16(false, a1, false, bf,
                                                   (short)0, acc1, false, false);
    bptr += bstep;
  }
  float* crow0 = C + (size_t)(mt0 * 16 + grp * 8) * ldc + nt * 16 + mlane;
  float* crow1 = crow0 + (size_t)16 * ldc;
#pragma unroll
  for (int r = 0; r < 8; ++r) crow0[(size_t)r * ldc] = acc0[r];
#pragma unroll
  for (int r = 0; r < 8; ++r) crow1[(size_t)r * ldc] = acc1[r];
}

// ---------------- attention scores a_src/a_dst ----------------
__global__ void attn_scores_k(const float* __restrict__ h, const float* __restrict__ att_s,
                              const float* __restrict__ att_d, float* as_o, float* ad_o,
                              int H, int C) {
  int i = blockIdx.x * blockDim.x + threadIdx.x;      // node*H + head
  if (i >= NN * H) return;
  int node = i / H, head = i % H;
  const float* hp = h + (size_t)node * H * C + (size_t)head * C;
  const float* s = att_s + head * C;
  const float* d = att_d + head * C;
  float a = 0.f, b = 0.f;
  for (int c = 0; c < C; ++c) { float v = hp[c]; a += v * s[c]; b += v * d[c]; }
  as_o[i] = a; ad_o[i] = b;
}

// ---------------- edge pass 1: segment max (encoded atomicMax) ----------------
__global__ void edge_max_k(const int* ei, int E, const float* asrc, const float* adst,
                           unsigned* m_enc, int H) {
  long i = (long)blockIdx.x * blockDim.x + threadIdx.x;
  long tot = (long)(E + NN) * H;
  if (i >= tot) return;
  int e = (int)(i / H), head = (int)(i % H);
  int s, d; get_edge(ei, E, e, s, d);
  float ev = asrc[s * H + head] + adst[d * H + head];
  ev = (ev >= 0.f) ? ev : NEG_SLOPE * ev;
  atomicMax(&m_enc[d * H + head], enc_f32(ev));
}

// ---------------- edge pass 2: denom = segment_sum(exp(e - m)) ----------------
__global__ void edge_den_k(const int* ei, int E, const float* asrc, const float* adst,
                           const unsigned* m_enc, float* den, int H) {
  long i = (long)blockIdx.x * blockDim.x + threadIdx.x;
  long tot = (long)(E + NN) * H;
  if (i >= tot) return;
  int e = (int)(i / H), head = (int)(i % H);
  int s, d; get_edge(ei, E, e, s, d);
  float ev = asrc[s * H + head] + adst[d * H + head];
  ev = (ev >= 0.f) ? ev : NEG_SLOPE * ev;
  float m = dec_f32(m_enc[d * H + head]);
  atomicAdd(&den[d * H + head], __expf(ev - m));
}

// ---------------- edge pass 3: scatter msg = h[src]*alpha into out[dst] ----------------
__global__ void edge_scatter_k(const int* ei, int E, const float* __restrict__ h,
                               const float* asrc, const float* adst,
                               const unsigned* m_enc, const float* den,
                               float* out, int H, int C) {
  int e = blockIdx.x;
  int s, d; get_edge(ei, E, e, s, d);
  int F = H * C;
  for (int j = threadIdx.x; j < F; j += blockDim.x) {
    int head = j / C;
    float ev = asrc[s * H + head] + adst[d * H + head];
    ev = (ev >= 0.f) ? ev : NEG_SLOPE * ev;
    float ex = __expf(ev - dec_f32(m_enc[d * H + head]));
    float alpha = ex / (den[d * H + head] + 1e-16f);
    atomicAdd(&out[(size_t)d * F + j], h[(size_t)s * F + j] * alpha);
  }
}

// ---------------- bias + ELU -> bf16 (feeds GEMM2) ----------------
__global__ void elu_bias_bf16_k(const float* __restrict__ in, const float* b,
                                unsigned short* out, int F) {
  long i = (long)blockIdx.x * blockDim.x + threadIdx.x;
  if (i >= (long)NN * F) return;
  float v = in[i] + b[i % F];
  v = (v > 0.f) ? v : (__expf(v) - 1.f);
  out[i] = f32_to_bf16(v);
}

// ---------------- bias + ELU in-place f32 ----------------
__global__ void elu_bias_f32_k(float* x, const float* b, int F) {
  long i = (long)blockIdx.x * blockDim.x + threadIdx.x;
  if (i >= (long)NN * F) return;
  float v = x[i] + b[i % F];
  x[i] = (v > 0.f) ? v : (__expf(v) - 1.f);
}

// ---------------- global max pool (encoded atomicMax) ----------------
__global__ void pool_max_k(const float* __restrict__ h, const int* batch, unsigned* pool) {
  long i = (long)blockIdx.x * blockDim.x + threadIdx.x;
  if (i >= (long)NN * CC) return;
  int n = (int)(i / CC), c = (int)(i % CC);
  atomicMax(&pool[batch[n] * CC + c], enc_f32(h[i]));
}

// ---------------- FC + ReLU ----------------
__global__ void fc_relu_k(const unsigned* __restrict__ pool, const float* __restrict__ W,
                          const float* __restrict__ b, float* out) {
  __shared__ float g[CC];
  int gr = blockIdx.x, c = threadIdx.x;
  g[c] = dec_f32(pool[gr * CC + c]);
  __syncthreads();
  float acc = b[c];
  for (int k = 0; k < CC; ++k) acc += g[k] * W[k * CC + c];
  out[gr * CC + c] = (acc > 0.f) ? acc : 0.f;
}

// ---------------- workspace layout (bytes) ----------------
#define OFF_XPAD   0UL                               // 20000*96*2   = 3,840,000
#define OFF_WP1    3840000UL                         // 96*1280*2   =   245,760
#define OFF_WP2    4085760UL                         // 1280*128*2  =   327,680
#define OFF_H1     4413440UL                         // 20000*1280*4= 102,400,000 (reused: bf16 h1act)
#define OFF_ASRC1  106813440UL                       // 20000*10*4
#define OFF_ADST1  107613440UL
#define OFF_M1     108413440UL
#define OFF_DEN1   109213440UL
#define OFF_OUT1   110013440UL                       // 102,400,000
#define OFF_H2     212413440UL                       // 20000*128*4
#define OFF_ASRC2  222653440UL
#define OFF_ADST2  222733440UL
#define OFF_M2     222813440UL
#define OFF_DEN2   222893440UL
#define OFF_OUT2   222973440UL                       // 10,240,000
#define OFF_POOL   233213440UL                       // 128*128*4

extern "C" void kernel_launch(void* const* d_in, const int* in_sizes, int n_in,
                              void* d_out, int out_size, void* d_ws, size_t ws_size,
                              hipStream_t stream) {
  const float* x        = (const float*)d_in[0];
  const int*   ei       = (const int*)  d_in[1];
  const int*   batch    = (const int*)  d_in[2];
  const float* W1       = (const float*)d_in[3];
  const float* att_src1 = (const float*)d_in[4];
  const float* att_dst1 = (const float*)d_in[5];
  const float* b1       = (const float*)d_in[6];
  const float* W2       = (const float*)d_in[7];
  const float* att_src2 = (const float*)d_in[8];
  const float* att_dst2 = (const float*)d_in[9];
  const float* b2       = (const float*)d_in[10];
  const float* fc_W     = (const float*)d_in[11];
  const float* fc_b     = (const float*)d_in[12];
  float* out = (float*)d_out;

  const int E = in_sizes[1] / 2;            // 320000
  char* ws = (char*)d_ws;
  unsigned short* xpad = (unsigned short*)(ws + OFF_XPAD);
  unsigned short* wp1  = (unsigned short*)(ws + OFF_WP1);
  unsigned short* wp2  = (unsigned short*)(ws + OFF_WP2);
  float*    h1    = (float*)(ws + OFF_H1);
  unsigned short* h1act = (unsigned short*)(ws + OFF_H1);   // reuse after layer-1 scatter
  float*    asrc1 = (float*)(ws + OFF_ASRC1);
  float*    adst1 = (float*)(ws + OFF_ADST1);
  unsigned* m1    = (unsigned*)(ws + OFF_M1);
  float*    den1  = (float*)(ws + OFF_DEN1);
  float*    out1  = (float*)(ws + OFF_OUT1);
  float*    h2    = (float*)(ws + OFF_H2);
  float*    asrc2 = (float*)(ws + OFF_ASRC2);
  float*    adst2 = (float*)(ws + OFF_ADST2);
  unsigned* m2    = (unsigned*)(ws + OFF_M2);
  float*    den2  = (float*)(ws + OFF_DEN2);
  float*    out2  = (float*)(ws + OFF_OUT2);
  unsigned* pool  = (unsigned*)(ws + OFF_POOL);

  const int B = 256;
  auto blocks = [](long n, int b) { return (unsigned)((n + b - 1) / b); };

  // ---- init accumulators ----
  fill_u32_k<<<blocks((long)NN * F1, B), B, 0, stream>>>((unsigned*)out1, 0u, (long)NN * F1);
  fill_u32_k<<<blocks((long)NN * H1, B), B, 0, stream>>>((unsigned*)den1, 0u, (long)NN * H1);
  fill_u32_k<<<blocks((long)NN * H1, B), B, 0, stream>>>(m1, ENC_NEG_INF, (long)NN * H1);
  fill_u32_k<<<blocks((long)NN * CC, B), B, 0, stream>>>((unsigned*)out2, 0u, (long)NN * CC);
  fill_u32_k<<<blocks(NN, B), B, 0, stream>>>((unsigned*)den2, 0u, NN);
  fill_u32_k<<<blocks(NN, B), B, 0, stream>>>(m2, ENC_NEG_INF, NN);
  fill_u32_k<<<blocks(GG * CC, B), B, 0, stream>>>(pool, ENC_NEG_INF, GG * CC);

  // ---- convert / pack ----
  cvt_x_k<<<blocks((long)NN * K1P, B), B, 0, stream>>>(x, xpad);
  pack_w_k<<<blocks((long)K1P * F1, B), B, 0, stream>>>(W1, wp1, F_IN, K1P, F1);
  pack_w_k<<<blocks((long)F1 * CC, B), B, 0, stream>>>(W2, wp2, F1, F1, CC);

  // ---- layer 1: GEMM (WMMA bf16, 2 M-tiles/wave) ----
  gemm_bf16_wmma_k<<<dim3(NN / 32, F1 / 16 / 8), B, 0, stream>>>(xpad, wp1, h1, K1P, F1, F1 / 16, K1P / 32);
  attn_scores_k<<<blocks((long)NN * H1, B), B, 0, stream>>>(h1, att_src1, att_dst1, asrc1, adst1, H1, CC);
  edge_max_k<<<blocks((long)(E + NN) * H1, B), B, 0, stream>>>(ei, E, asrc1, adst1, m1, H1);
  edge_den_k<<<blocks((long)(E + NN) * H1, B), B, 0, stream>>>(ei, E, asrc1, adst1, m1, den1, H1);
  edge_scatter_k<<<E + NN, 256, 0, stream>>>(ei, E, h1, asrc1, adst1, m1, den1, out1, H1, CC);
  elu_bias_bf16_k<<<blocks((long)NN * F1, B), B, 0, stream>>>(out1, b1, h1act, F1);

  // ---- layer 2: GEMM (WMMA bf16, 2 M-tiles/wave) ----
  gemm_bf16_wmma_k<<<dim3(NN / 32, 1), B, 0, stream>>>(h1act, wp2, h2, F1, CC, CC / 16, F1 / 32);
  attn_scores_k<<<blocks(NN, B), B, 0, stream>>>(h2, att_src2, att_dst2, asrc2, adst2, 1, CC);
  edge_max_k<<<blocks((long)(E + NN), B), B, 0, stream>>>(ei, E, asrc2, adst2, m2, 1);
  edge_den_k<<<blocks((long)(E + NN), B), B, 0, stream>>>(ei, E, asrc2, adst2, m2, den2, 1);
  edge_scatter_k<<<E + NN, 128, 0, stream>>>(ei, E, h2, asrc2, adst2, m2, den2, out2, 1, CC);
  elu_bias_f32_k<<<blocks((long)NN * CC, B), B, 0, stream>>>(out2, b2, CC);

  // ---- pool + FC ----
  pool_max_k<<<blocks((long)NN * CC, B), B, 0, stream>>>(out2, batch, pool);
  fc_relu_k<<<GG, CC, 0, stream>>>(pool, fc_W, fc_b, out);
}
